// PointNetLayer_88012469830598
// MI455X (gfx1250) — matrix operand order repaired and checked by reading, not compiled, not run on hardware
//
#include <hip/hip_runtime.h>

#define N_NODES 100000
#define N_EDGES 1600000
#define INC 16
#define H1 64
#define H2 128
#define OUTC 64

typedef __attribute__((ext_vector_type(16))) _Float16 v16h;
typedef __attribute__((ext_vector_type(8)))  _Float16 v8h;
typedef __attribute__((ext_vector_type(8)))  float    v8f;
typedef __attribute__((ext_vector_type(4)))  float    v4f;

static __device__ __forceinline__ v8f wmma_f16(v16h a, v16h b, v8f c) {
  // D = A(16x32 f16) * B(32x16 f16) + C(16x16 f32)
  return __builtin_amdgcn_wmma_f32_16x16x32_f16(false, a, false, b, (short)0, c, false, false);
}

static __device__ __forceinline__ v8f bcast8(float v) {
  v8f r;
#pragma unroll
  for (int i = 0; i < 8; ++i) r[i] = v;
  return r;
}

static __device__ __forceinline__ v16h cat8(v8h lo, v8h hi) {
  v16h r;
#pragma unroll
  for (int i = 0; i < 8; ++i) { r[i] = lo[i]; r[8 + i] = hi[i]; }
  return r;
}

// relu as a single v_med3; clamp to f16-max when feeding an f16 convert
static __device__ __forceinline__ float relu_h(float x) {
  return __builtin_amdgcn_fmed3f(x, 0.0f, 65504.0f);
}
static __device__ __forceinline__ float relu_f(float x) {
  return __builtin_amdgcn_fmed3f(x, 0.0f, 3.4028234663852886e38f);
}

__global__ void zero_f32(float4* __restrict__ p, int n4) {
  int i = blockIdx.x * blockDim.x + threadIdx.x;
  int stride = gridDim.x * blockDim.x;
  float4 z = make_float4(0.f, 0.f, 0.f, 0.f);
  for (; i < n4; i += stride) p[i] = z;
}

// Per-edge MLP (19->64->128->64, ReLU) + scatter max into agg (u32 bit-pattern max).
__global__ __launch_bounds__(128) void edge_mlp(
    const float* __restrict__ x, const float* __restrict__ pos,
    const long long* __restrict__ ei,
    const float* __restrict__ w1, const float* __restrict__ b1,
    const float* __restrict__ w2, const float* __restrict__ b2,
    const float* __restrict__ w3, const float* __restrict__ b3,
    unsigned int* __restrict__ agg)
{
  __shared__ __align__(16) _Float16 sW1[32 * H1];     // K padded 19->32; row 19 = bias
  __shared__ __align__(16) _Float16 sW2[H1 * H2];
  __shared__ __align__(16) _Float16 sW3[H2 * OUTC];
  __shared__ float sB2[H2];
  __shared__ float sB3[OUTC];
  __shared__ __align__(16) _Float16 sHa[4][16 * H1];  // per-wave h1 (16x64)
  __shared__ __align__(16) _Float16 sHb[4][16 * H2];  // per-wave h2 (16x128)

  for (int i = threadIdx.x; i < 32 * H1; i += blockDim.x) {
    int k = i >> 6, n = i & 63;
    _Float16 v = (_Float16)0.0f;
    if (k < INC + 3) v = (_Float16)w1[k * H1 + n];
    else if (k == INC + 3) v = (_Float16)b1[n];   // bias row, paired with A's K19 == 1.0
    sW1[i] = v;
  }
  for (int i = threadIdx.x; i < H1 * H2; i += blockDim.x) sW2[i] = (_Float16)w2[i];
  for (int i = threadIdx.x; i < H2 * OUTC; i += blockDim.x) sW3[i] = (_Float16)w3[i];
  if (threadIdx.x < H2)   sB2[threadIdx.x] = b2[threadIdx.x];
  if (threadIdx.x < OUTC) sB3[threadIdx.x] = b3[threadIdx.x];
  __syncthreads();

  const int lane = threadIdx.x & 31;
  const int w    = threadIdx.x >> 5;
  const int ln   = lane & 15;
  const bool hi  = lane >= 16;
  const int m0   = hi ? 8 : 0;
  const int gwave  = blockIdx.x * (blockDim.x >> 5) + w;
  const int nwaves = gridDim.x * (blockDim.x >> 5);
  _Float16* __restrict__ h1 = sHa[w];
  _Float16* __restrict__ h2 = sHb[w];

  for (int tile = gwave; tile < N_EDGES / 16; tile += nwaves) {
    const int ebase = tile << 4;
    const int e = ebase + ln;                       // this lane's edge row
    const int src  = (int)ei[e];
    const int dsti = (int)ei[(size_t)N_EDGES + e];

    // ---- A0 fragment: msg = [x_src(16) | rel_pos(3) | 1.0 | pad] as 16x32 f16 ----
    // lanes 0-15: K0-7 (elts 0-7), K16-23 (elts 8-15); lanes 16-31: K8-15, K24-31
    v16h a0;
    {
      const v4f* xp = (const v4f*)(x + (size_t)src * INC + (hi ? 8 : 0));
      v4f x0 = xp[0], x1 = xp[1];
#pragma unroll
      for (int i = 0; i < 4; ++i) { a0[i] = (_Float16)x0[i]; a0[4 + i] = (_Float16)x1[i]; }
#pragma unroll
      for (int i = 8; i < 16; ++i) a0[i] = (_Float16)0.0f;
      if (!hi) {  // K16..18 = rel_pos, K19 = 1.0 (bias row selector)
        a0[8]  = (_Float16)(pos[src * 3 + 0] - pos[dsti * 3 + 0]);
        a0[9]  = (_Float16)(pos[src * 3 + 1] - pos[dsti * 3 + 1]);
        a0[10] = (_Float16)(pos[src * 3 + 2] - pos[dsti * 3 + 2]);
        a0[11] = (_Float16)1.0f;
      }
    }

    // ---- layer 1: [16x32] * [32x64], bias via K row -> C starts at inline 0 ----
#pragma unroll
    for (int t = 0; t < 4; ++t) {
      v16h b; for (int s = 0; s < 16; ++s) b[s] = sW1[(m0 + s + (s >= 8 ? 8 : 0)) * H1 + t * 16 + ln];      // lane = N col, elts = K (table order)
      v8f c = {};                                           // SRC2 = inline 0
      c = wmma_f16(a0, b, c);
#pragma unroll
      for (int g = 0; g < 8; ++g)
        h1[(m0 + g) * H1 + t * 16 + ln] = (_Float16)relu_h(c[g]);
    }
    asm volatile("s_wait_dscnt 0" ::: "memory");

    // ---- layer 2: [16x64] * [64x128] ----
    v16h a1[2];
#pragma unroll
    for (int kc = 0; kc < 2; ++kc) {
      int base = ln * H1 + kc * 32 + (hi ? 8 : 0);
      a1[kc] = cat8(*(const v8h*)&h1[base], *(const v8h*)&h1[base + 16]);
    }
#pragma unroll
    for (int t = 0; t < 8; ++t) {
      v8f c = bcast8(sB2[t * 16 + ln]);
#pragma unroll
      for (int kc = 0; kc < 2; ++kc) {
        v16h b; for (int s = 0; s < 16; ++s) b[s] = sW2[(kc * 32 + m0 + s + (s >= 8 ? 8 : 0)) * H2 + t * 16 + ln];
        c = wmma_f16(a1[kc], b, c);
      }
#pragma unroll
      for (int g = 0; g < 8; ++g)
        h2[(m0 + g) * H2 + t * 16 + ln] = (_Float16)relu_h(c[g]);
    }
    asm volatile("s_wait_dscnt 0" ::: "memory");

    // ---- layer 3: [16x128] * [128x64] + scatter atomic max ----
    v16h a2[4];
#pragma unroll
    for (int kc = 0; kc < 4; ++kc) {
      int base = ln * H2 + kc * 32 + (hi ? 8 : 0);
      a2[kc] = cat8(*(const v8h*)&h2[base], *(const v8h*)&h2[base + 16]);
    }
    int drow[8];
#pragma unroll
    for (int g = 0; g < 8; ++g) drow[g] = __shfl(dsti, m0 + g, 32);
#pragma unroll
    for (int t = 0; t < 4; ++t) {
      v8f c = bcast8(sB3[t * 16 + ln]);
#pragma unroll
      for (int kc = 0; kc < 4; ++kc) {
        v16h b; for (int s = 0; s < 16; ++s) b[s] = sW3[(kc * 32 + m0 + s + (s >= 8 ? 8 : 0)) * OUTC + t * 16 + ln];
        c = wmma_f16(a2[kc], b, c);
      }
#pragma unroll
      for (int g = 0; g < 8; ++g) {
        unsigned int v = __float_as_uint(relu_f(c[g]));  // relu >= 0 -> u32 max == f32 max
        atomicMax(agg + (size_t)drow[g] * OUTC + t * 16 + ln, v);
      }
    }
  }
}

// out = relu(agg @ wg + bg), in-place on the agg buffer.
__global__ __launch_bounds__(256) void node_mlp(
    const float* __restrict__ wg, const float* __restrict__ bg,
    float* __restrict__ io)
{
  __shared__ __align__(16) _Float16 sWg[OUTC * OUTC];
  __shared__ float sBg[OUTC];
  for (int i = threadIdx.x; i < OUTC * OUTC; i += blockDim.x) sWg[i] = (_Float16)wg[i];
  if (threadIdx.x < OUTC) sBg[threadIdx.x] = bg[threadIdx.x];
  __syncthreads();

  const int lane = threadIdx.x & 31;
  const int ln   = lane & 15;
  const bool hi  = lane >= 16;
  const int m0   = hi ? 8 : 0;
  const int gwave  = blockIdx.x * (blockDim.x >> 5) + (threadIdx.x >> 5);
  const int nwaves = gridDim.x * (blockDim.x >> 5);

  for (int tile = gwave; tile < N_NODES / 16; tile += nwaves) {
    const int nb = tile << 4;
    const float* ar = io + (size_t)(nb + ln) * OUTC;
    v16h a[2];
#pragma unroll
    for (int kc = 0; kc < 2; ++kc) {
      int k0 = kc * 32 + (hi ? 8 : 0);
      v4f f0 = *(const v4f*)(ar + k0);
      v4f f1 = *(const v4f*)(ar + k0 + 4);
      v4f f2 = *(const v4f*)(ar + k0 + 16);
      v4f f3 = *(const v4f*)(ar + k0 + 20);
#pragma unroll
      for (int i = 0; i < 4; ++i) {
        a[kc][i]      = (_Float16)f0[i];
        a[kc][4 + i]  = (_Float16)f1[i];
        a[kc][8 + i]  = (_Float16)f2[i];
        a[kc][12 + i] = (_Float16)f3[i];
      }
    }
#pragma unroll
    for (int t = 0; t < 4; ++t) {
      v8f c = bcast8(sBg[t * 16 + ln]);
#pragma unroll
      for (int kc = 0; kc < 2; ++kc) {
        v16h b; for (int s = 0; s < 16; ++s) b[s] = sWg[(kc * 32 + m0 + s + (s >= 8 ? 8 : 0)) * OUTC + t * 16 + ln];
        c = wmma_f16(a[kc], b, c);
      }
#pragma unroll
      for (int g = 0; g < 8; ++g)
        io[(size_t)(nb + m0 + g) * OUTC + t * 16 + ln] = relu_f(c[g]);
    }
  }
}

extern "C" void kernel_launch(void* const* d_in, const int* in_sizes, int n_in,
                              void* d_out, int out_size, void* d_ws, size_t ws_size,
                              hipStream_t stream) {
  (void)in_sizes; (void)n_in; (void)d_ws; (void)ws_size;
  const float* x   = (const float*)d_in[0];
  const float* pos = (const float*)d_in[1];
  const float* w1  = (const float*)d_in[2];
  const float* b1  = (const float*)d_in[3];
  const float* w2  = (const float*)d_in[4];
  const float* b2  = (const float*)d_in[5];
  const float* w3  = (const float*)d_in[6];
  const float* b3  = (const float*)d_in[7];
  const float* wg  = (const float*)d_in[8];
  const float* bg  = (const float*)d_in[9];
  const long long* ei = (const long long*)d_in[10];
  float* out = (float*)d_out;

  // 1) agg = 0 (exactly reproduces reference: relu>=0 and isolated nodes -> 0)
  zero_f32<<<1024, 256, 0, stream>>>((float4*)out, out_size / 4);
  // 2) per-edge MLP + atomic segment-max into out
  edge_mlp<<<2048, 128, 0, stream>>>(x, pos, ei, w1, b1, w2, b2, w3, b3,
                                     (unsigned int*)out);
  // 3) node MLP, in place
  node_mlp<<<512, 256, 0, stream>>>(wg, bg, out);
}
